// MultiheadAttention_40020505264268
// MI455X (gfx1250) — matrix operand order repaired and checked by reading, not compiled, not run on hardware
//
#include <hip/hip_runtime.h>
#include <stdint.h>

#define D_  1024
#define H_  16
#define DH_ 64
#define S_  2048
#define B_  4

typedef __attribute__((ext_vector_type(16))) __bf16 v16bf;
typedef __attribute__((ext_vector_type(8)))  __bf16 v8bf;
typedef __attribute__((ext_vector_type(8)))  float  v8f;

// ---------------- fragment loaders ----------------
// A-fragment 16(M)x32(K) bf16: lane l -> row m = l&15,
// element e -> k = (e&7) + 8*(l>>4) + 16*(e>>3). Two contiguous 16B chunks.
__device__ __forceinline__ v16bf load_frag_a(const __bf16* p, int ld, int lane) {
    const __bf16* rp = p + (lane & 15) * ld + ((lane >> 4) << 3);
    v8bf lo = *(const v8bf*)(rp);
    v8bf hi = *(const v8bf*)(rp + 16);
    return __builtin_shufflevector(lo, hi, 0,1,2,3,4,5,6,7,8,9,10,11,12,13,14,15);
}

// B-fragment 32(K)x16(N) bf16 stored column-major (each "row" of p is one N
// column, K contiguous): lane l -> col n = l&15, element e -> k = (e&7) + 8*(l>>4) + 16*(e>>3).
__device__ __forceinline__ v16bf load_frag_b(const __bf16* p, int ld, int lane) {
    const __bf16* rp = p + (lane & 15) * ld + ((lane >> 4) << 3);
    v8bf lo = *(const v8bf*)(rp);
    v8bf hi = *(const v8bf*)(rp + 16);
    return __builtin_shufflevector(lo, hi, 0,1,2,3,4,5,6,7,8,9,10,11,12,13,14,15);
}

__device__ __forceinline__ v8f wmma_bf16(v16bf a, v16bf b, v8f c) {
    return __builtin_amdgcn_wmma_f32_16x16x32_bf16(
        false, a, false, b, (short)0, c, false, false);
}

// CDNA5 async global -> LDS copy (ASYNCcnt-tracked, no VGPR staging).
// lds_off: wave-relative LDS byte address (generic LDS ptr truncated to 32b).
__device__ __forceinline__ void async_b128(unsigned lds_off, const __bf16* gaddr) {
    asm volatile("global_load_async_to_lds_b128 %0, %1, off"
                 :: "v"(lds_off), "v"(gaddr)
                 : "memory");
}

// ---------------- conversion / transpose ----------------
__global__ void cvt_bf16(const float* __restrict__ in, __bf16* __restrict__ out, int n) {
    int i = blockIdx.x * blockDim.x + threadIdx.x;
    if (i < n) out[i] = (__bf16)in[i];
}

// W: [K=D_, N=D_] row-major f32  ->  Wt: [N][K] bf16
__global__ void transpose_bf16(const float* __restrict__ W, __bf16* __restrict__ Wt) {
    int k = blockIdx.x * 256 + threadIdx.x;
    int n = blockIdx.y;
    Wt[(size_t)n * D_ + k] = (__bf16)W[(size_t)k * D_ + n];
}

// ---------------- QKV projection GEMM ----------------
// C[M,N] = A[M,K] * Wt[N,K]^T + bias ; per-wave 32x64 tile, 4 waves/block
// (block tile 128x64). The 64x32 weight tile is staged per block into
// double-buffered LDS with global_load_async_to_lds_b128 (ASYNCcnt), so the
// 4 waves share one copy; B-fragments come from LDS (ds_load_b128).
// outN: normal bf16 [M, D_]; outT: per-head transposed bf16 [B_,H_,DH_,S_].
#define LDB 40          // LDS row pitch in elements (80B: 16B aligned, bank-spread)
#define LBUF_HALF (64 * LDB)
__global__ __launch_bounds__(128) void gemm_qkv(
    const __bf16* __restrict__ A, const __bf16* __restrict__ Wt,
    const float* __restrict__ bias,
    __bf16* __restrict__ outN, __bf16* __restrict__ outT)
{
    __shared__ __bf16 lbuf[2][LBUF_HALF];
    int tid = threadIdx.x;
    int wave = tid >> 5, lane = tid & 31;
    int m0 = (blockIdx.x * 4 + wave) * 32;
    int n0 = blockIdx.y * 64;
    int row = (lane >> 4) << 3, col = lane & 15;

    const __bf16* ap0 = A + (size_t)m0 * D_;
    const __bf16* ap1 = ap0 + (size_t)16 * D_;
    const __bf16* bp  = Wt + (size_t)n0 * D_;

    // 256 16B chunks per 64x32 tile; 2 per thread. chunk c: row n=c>>2, q=c&3.
    int c0 = tid * 2, c1 = c0 + 1;
    int na = c0 >> 2, qa = c0 & 3;
    int nb = c1 >> 2, qb = c1 & 3;
    unsigned lds0 = (unsigned)(uintptr_t)(const void*)&lbuf[0][0];
    unsigned la = lds0 + (unsigned)(na * (LDB * 2) + qa * 16);
    unsigned lc = lds0 + (unsigned)(nb * (LDB * 2) + qb * 16);
    const __bf16* ga = bp + (size_t)na * D_ + qa * 8;
    const __bf16* gb = bp + (size_t)nb * D_ + qb * 8;

    // prologue: stage k=0 into buffer 0
    async_b128(la, ga);
    async_b128(lc, gb);

    v8f acc[2][4] = {};
    for (int k = 0; k < D_; k += 32) {
        int t = k >> 5;
        unsigned coff = (t & 1) ? (unsigned)(LBUF_HALF * 2) : 0u;  // bytes
        unsigned noff = (t & 1) ? 0u : (unsigned)(LBUF_HALF * 2);
        __syncthreads();  // all waves done reading the buffer we overwrite next
        if (k + 32 < D_) {
            async_b128(la + noff, ga + k + 32);
            async_b128(lc + noff, gb + k + 32);
        }
        // A fragments (global); stay in flight across the waits below.
        v16bf a0 = load_frag_a(ap0 + k, D_, lane);
        v16bf a1 = load_frag_a(ap1 + k, D_, lane);
        if (k + 32 < D_) asm volatile("s_wait_asynccnt 0x2" ::: "memory");
        else             asm volatile("s_wait_asynccnt 0x0" ::: "memory");
        __syncthreads();  // whole current tile staged in LDS
        const __bf16* lbase = (const __bf16*)&lbuf[0][0] + (coff >> 1);
        v16bf b0 = load_frag_b(lbase,            LDB, lane);
        v16bf b1 = load_frag_b(lbase + 16 * LDB, LDB, lane);
        v16bf b2 = load_frag_b(lbase + 32 * LDB, LDB, lane);
        v16bf b3 = load_frag_b(lbase + 48 * LDB, LDB, lane);
        acc[0][0] = wmma_bf16(a0, b0, acc[0][0]);
        acc[0][1] = wmma_bf16(a0, b1, acc[0][1]);
        acc[0][2] = wmma_bf16(a0, b2, acc[0][2]);
        acc[0][3] = wmma_bf16(a0, b3, acc[0][3]);
        acc[1][0] = wmma_bf16(a1, b0, acc[1][0]);
        acc[1][1] = wmma_bf16(a1, b1, acc[1][1]);
        acc[1][2] = wmma_bf16(a1, b2, acc[1][2]);
        acc[1][3] = wmma_bf16(a1, b3, acc[1][3]);
    }

#pragma unroll
    for (int mt = 0; mt < 2; ++mt) {
#pragma unroll
        for (int j = 0; j < 4; ++j) {
            int n = n0 + j * 16 + col;
            float bv = bias[n];
#pragma unroll
            for (int r = 0; r < 8; ++r) {
                float val = acc[mt][j][r] + bv;
                int m = m0 + mt * 16 + row + r;
                if (outN) outN[(size_t)m * D_ + n] = (__bf16)val;
                if (outT) {
                    int bb = m >> 11, ss = m & (S_ - 1);
                    int hh = n >> 6, dh = n & 63;
                    outT[(((size_t)(bb * H_ + hh)) * DH_ + dh) * S_ + ss] = (__bf16)val;
                }
            }
        }
    }
}

// ---------------- flash attention ----------------
// One wave = one 16-query tile of one (b,h). Streams keys 32 at a time.
// All K and V fragments for the block are issued up front: score WMMAs wait
// only on K, V loads stay in flight across the softmax VALU work.
__global__ __launch_bounds__(128) void attn_kernel(
    const __bf16* __restrict__ qbf, const __bf16* __restrict__ kbf,
    const __bf16* __restrict__ vt,  const long long* __restrict__ ids,
    __bf16* __restrict__ attn)
{
    __shared__ __bf16 pshm[4][16 * 40];  // per-wave P staging, ld=40 (16B-aligned rows)
    int wave = threadIdx.x >> 5, lane = threadIdx.x & 31;
    int b = blockIdx.z, h = blockIdx.y;
    int m0 = blockIdx.x * 64 + wave * 16;
    int row = (lane >> 4) << 3, col = lane & 15;

    const __bf16* qp = qbf + (size_t)(b * S_ + m0) * D_ + h * DH_;
    v16bf qa0 = load_frag_a(qp, D_, lane);
    v16bf qa1 = load_frag_a(qp + 32, D_, lane);

    v8f o0 = {}, o1 = {}, o2 = {}, o3 = {};
    float rm[8], rl[8];
#pragma unroll
    for (int r = 0; r < 8; ++r) { rm[r] = -1e30f; rl[r] = 0.0f; }

    const __bf16* kbase = kbf + (size_t)b * S_ * D_ + h * DH_;
    const __bf16* vbase = vt + ((size_t)(b * H_ + h)) * DH_ * S_;
    const long long* idb = ids + (size_t)b * S_;
    __bf16* pl = &pshm[wave][0];

    for (int kb = 0; kb < S_; kb += 32) {
        // Issue every load for this block first (K frags, then V frags).
        v16bf kf00 = load_frag_b(kbase + (size_t)kb * D_,             D_, lane);
        v16bf kf01 = load_frag_b(kbase + (size_t)kb * D_ + 32,        D_, lane);
        v16bf kf10 = load_frag_b(kbase + (size_t)(kb + 16) * D_,      D_, lane);
        v16bf kf11 = load_frag_b(kbase + (size_t)(kb + 16) * D_ + 32, D_, lane);
        v16bf vf0 = load_frag_b(vbase + (size_t)0  * S_ + kb, S_, lane);
        v16bf vf1 = load_frag_b(vbase + (size_t)16 * S_ + kb, S_, lane);
        v16bf vf2 = load_frag_b(vbase + (size_t)32 * S_ + kb, S_, lane);
        v16bf vf3 = load_frag_b(vbase + (size_t)48 * S_ + kb, S_, lane);

        // Prefetch next key block (K rows + V rows) into cache.
        int kbn = (kb + 32 < S_) ? kb + 32 : kb;
        __builtin_prefetch(kbase + (size_t)(kbn + col) * D_, 0, 3);
        __builtin_prefetch(kbase + (size_t)(kbn + 16 + col) * D_, 0, 3);
        __builtin_prefetch(vbase + (size_t)(lane * 2) * S_ + kbn, 0, 3);
        __builtin_prefetch(vbase + (size_t)(lane * 2 + 1) * S_ + kbn, 0, 3);

        v8f s0 = {}, s1 = {};
        s0 = wmma_bf16(qa0, kf00, s0);
        s0 = wmma_bf16(qa1, kf01, s0);
        s1 = wmma_bf16(qa0, kf10, s1);
        s1 = wmma_bf16(qa1, kf11, s1);

        float mask0 = (idb[kb + col]      != 0) ? 0.0f : -1e30f;
        float mask1 = (idb[kb + 16 + col] != 0) ? 0.0f : -1e30f;

        float p0[8], p1[8];
#pragma unroll
        for (int r = 0; r < 8; ++r) {
            float a = s0[r] * 0.125f + mask0;
            float c = s1[r] * 0.125f + mask1;
            float v = fmaxf(a, c);
#pragma unroll
            for (int off = 8; off; off >>= 1) v = fmaxf(v, __shfl_xor(v, off, 16));
            float nm = fmaxf(rm[r], v);
            float scale = __expf(rm[r] - nm);
            rm[r] = nm;
            float e0 = __expf(a - nm), e1 = __expf(c - nm);
            p0[r] = e0; p1[r] = e1;
            float ps = e0 + e1;
#pragma unroll
            for (int off = 8; off; off >>= 1) ps += __shfl_xor(ps, off, 16);
            rl[r] = rl[r] * scale + ps;
            o0[r] *= scale; o1[r] *= scale; o2[r] *= scale; o3[r] *= scale;
        }
        // C-layout -> row-major 16x32 P tile in this wave's private LDS slab.
        // LDS ops from one wave execute in order, so no block barrier needed.
#pragma unroll
        for (int r = 0; r < 8; ++r) {
            pl[(row + r) * 40 + col]      = (__bf16)p0[r];
            pl[(row + r) * 40 + 16 + col] = (__bf16)p1[r];
        }
        asm volatile("s_wait_dscnt 0x0" ::: "memory");
        v16bf pa = load_frag_a(pl, 40, lane);

        o0 = wmma_bf16(pa, vf0, o0);
        o1 = wmma_bf16(pa, vf1, o1);
        o2 = wmma_bf16(pa, vf2, o2);
        o3 = wmma_bf16(pa, vf3, o3);
    }

    __bf16* outp = attn + (size_t)(b * S_ + m0) * D_ + h * DH_;
#pragma unroll
    for (int r = 0; r < 8; ++r) {
        float inv = 1.0f / rl[r];
        outp[(row + r) * D_ + col]      = (__bf16)(o0[r] * inv);
        outp[(row + r) * D_ + 16 + col] = (__bf16)(o1[r] * inv);
        outp[(row + r) * D_ + 32 + col] = (__bf16)(o2[r] * inv);
        outp[(row + r) * D_ + 48 + col] = (__bf16)(o3[r] * inv);
    }
}

// ---------------- output projection + residual ----------------
__global__ __launch_bounds__(128) void gemm_proj(
    const __bf16* __restrict__ A, const __bf16* __restrict__ Wt,
    const float* __restrict__ bias, const float* __restrict__ xres,
    float* __restrict__ out)
{
    __shared__ __bf16 lbuf[2][LBUF_HALF];
    int tid = threadIdx.x;
    int wave = tid >> 5, lane = tid & 31;
    int m0 = (blockIdx.x * 4 + wave) * 32;
    int n0 = blockIdx.y * 64;
    int row = (lane >> 4) << 3, col = lane & 15;

    const __bf16* ap0 = A + (size_t)m0 * D_;
    const __bf16* ap1 = ap0 + (size_t)16 * D_;
    const __bf16* bp  = Wt + (size_t)n0 * D_;

    int c0 = tid * 2, c1 = c0 + 1;
    int na = c0 >> 2, qa = c0 & 3;
    int nb = c1 >> 2, qb = c1 & 3;
    unsigned lds0 = (unsigned)(uintptr_t)(const void*)&lbuf[0][0];
    unsigned la = lds0 + (unsigned)(na * (LDB * 2) + qa * 16);
    unsigned lc = lds0 + (unsigned)(nb * (LDB * 2) + qb * 16);
    const __bf16* ga = bp + (size_t)na * D_ + qa * 8;
    const __bf16* gb = bp + (size_t)nb * D_ + qb * 8;

    async_b128(la, ga);
    async_b128(lc, gb);

    v8f acc[2][4] = {};
    for (int k = 0; k < D_; k += 32) {
        int t = k >> 5;
        unsigned coff = (t & 1) ? (unsigned)(LBUF_HALF * 2) : 0u;
        unsigned noff = (t & 1) ? 0u : (unsigned)(LBUF_HALF * 2);
        __syncthreads();
        if (k + 32 < D_) {
            async_b128(la + noff, ga + k + 32);
            async_b128(lc + noff, gb + k + 32);
        }
        v16bf a0 = load_frag_a(ap0 + k, D_, lane);
        v16bf a1 = load_frag_a(ap1 + k, D_, lane);
        if (k + 32 < D_) asm volatile("s_wait_asynccnt 0x2" ::: "memory");
        else             asm volatile("s_wait_asynccnt 0x0" ::: "memory");
        __syncthreads();
        const __bf16* lbase = (const __bf16*)&lbuf[0][0] + (coff >> 1);
        v16bf b0 = load_frag_b(lbase,            LDB, lane);
        v16bf b1 = load_frag_b(lbase + 16 * LDB, LDB, lane);
        v16bf b2 = load_frag_b(lbase + 32 * LDB, LDB, lane);
        v16bf b3 = load_frag_b(lbase + 48 * LDB, LDB, lane);
        acc[0][0] = wmma_bf16(a0, b0, acc[0][0]);
        acc[0][1] = wmma_bf16(a0, b1, acc[0][1]);
        acc[0][2] = wmma_bf16(a0, b2, acc[0][2]);
        acc[0][3] = wmma_bf16(a0, b3, acc[0][3]);
        acc[1][0] = wmma_bf16(a1, b0, acc[1][0]);
        acc[1][1] = wmma_bf16(a1, b1, acc[1][1]);
        acc[1][2] = wmma_bf16(a1, b2, acc[1][2]);
        acc[1][3] = wmma_bf16(a1, b3, acc[1][3]);
    }

#pragma unroll
    for (int mt = 0; mt < 2; ++mt) {
#pragma unroll
        for (int j = 0; j < 4; ++j) {
            int n = n0 + j * 16 + col;
            float bv = bias[n];
#pragma unroll
            for (int r = 0; r < 8; ++r) {
                size_t idx = (size_t)(m0 + mt * 16 + row + r) * D_ + n;
                out[idx] = acc[mt][j][r] + bv + xres[idx];
            }
        }
    }
}

// ---------------- launch ----------------
extern "C" void kernel_launch(void* const* d_in, const int* in_sizes, int n_in,
                              void* d_out, int out_size, void* d_ws, size_t ws_size,
                              hipStream_t stream) {
    const float*     x   = (const float*)d_in[0];
    const long long* ids = (const long long*)d_in[1];
    const float* Wq = (const float*)d_in[2];
    const float* bq = (const float*)d_in[3];
    const float* Wk = (const float*)d_in[4];
    const float* bk = (const float*)d_in[5];
    const float* Wv = (const float*)d_in[6];
    const float* bv = (const float*)d_in[7];
    const float* Wp = (const float*)d_in[8];
    const float* bp = (const float*)d_in[9];
    float* out = (float*)d_out;

    const size_t MB = 1024 * 1024;
    char* w = (char*)d_ws;
    __bf16* xbf   = (__bf16*)(w);             // 16 MB
    __bf16* qbf   = (__bf16*)(w + 16 * MB);   // 16 MB
    __bf16* kbf   = (__bf16*)(w + 32 * MB);   // 16 MB
    __bf16* vtb   = (__bf16*)(w + 48 * MB);   // 16 MB  [B,H,DH,S]
    __bf16* attnb = (__bf16*)(w + 64 * MB);   // 16 MB
    __bf16* wqt   = (__bf16*)(w + 80 * MB);   // 2 MB each
    __bf16* wkt   = (__bf16*)(w + 82 * MB);
    __bf16* wvt   = (__bf16*)(w + 84 * MB);
    __bf16* wpt   = (__bf16*)(w + 86 * MB);

    const int M = B_ * S_;  // 8192
    cvt_bf16<<<dim3((M * D_) / 256), 256, 0, stream>>>(x, xbf, M * D_);
    transpose_bf16<<<dim3(D_ / 256, D_), 256, 0, stream>>>(Wq, wqt);
    transpose_bf16<<<dim3(D_ / 256, D_), 256, 0, stream>>>(Wk, wkt);
    transpose_bf16<<<dim3(D_ / 256, D_), 256, 0, stream>>>(Wv, wvt);
    transpose_bf16<<<dim3(D_ / 256, D_), 256, 0, stream>>>(Wp, wpt);

    dim3 g1(M / 128, D_ / 64);
    gemm_qkv<<<g1, 128, 0, stream>>>(xbf, wqt, bq, qbf, (__bf16*)nullptr);
    gemm_qkv<<<g1, 128, 0, stream>>>(xbf, wkt, bk, kbf, (__bf16*)nullptr);
    gemm_qkv<<<g1, 128, 0, stream>>>(xbf, wvt, bv, (__bf16*)nullptr, vtb);

    dim3 g2(S_ / 64, H_, B_);
    attn_kernel<<<g2, 128, 0, stream>>>(qbf, kbf, vtb, ids, attnb);

    gemm_proj<<<g1, 128, 0, stream>>>(attnb, wpt, bp, x, out);
}
